// StructuralTransformer_16741782520606
// MI455X (gfx1250) — matrix operand order repaired and checked by reading, not compiled, not run on hardware
//
#include <hip/hip_runtime.h>
#include <hip/hip_bf16.h>

// ---------------------------------------------------------------------------
// StructuralTransformer on gfx1250 (CDNA5): f16 WMMA GEMMs (async-LDS double
// buffered) + flash attention with WMMA QK^T / PV.
// ---------------------------------------------------------------------------

typedef __attribute__((ext_vector_type(16))) _Float16 v16h;
typedef __attribute__((ext_vector_type(8)))  _Float16 v8h;
typedef __attribute__((ext_vector_type(4)))  _Float16 v4h;
typedef __attribute__((ext_vector_type(8)))  float    v8f;

#define BB     4
#define LSEQ   1024
#define DM     768
#define TD     2304      // 3*D
#define NH     8
#define HD     96
#define FF     2048
#define MROWS  (BB*LSEQ) // 4096
#define APAD   40        // padded LDS row stride in halves (80B, conflict-free)

__device__ __forceinline__ v16h cat8(v8h lo, v8h hi) {
    return __builtin_shufflevector(lo, hi, 0,1,2,3,4,5,6,7,8,9,10,11,12,13,14,15);
}

__device__ __forceinline__ v8f wmma_f16(v16h a, v16h b, v8f c) {
    // v_wmma_f32_16x16x32_f16 : D = A(16x32 f16) x B(32x16 f16) + C(16x16 f32)
    return __builtin_amdgcn_wmma_f32_16x16x32_f16(false, a, false, b, (short)0, c,
                                                  false, false);
}

// Async DMA of one 16B chunk into LDS (ASYNCcnt-tracked; per-lane addresses).
__device__ __forceinline__ void async_ldscopy_b128(const _Float16* g, void* lds) {
    uint32_t laddr = (uint32_t)(uintptr_t)lds;        // low 32 bits = LDS offset
    uint64_t gaddr = (uint64_t)(uintptr_t)g;
    asm volatile("global_load_async_to_lds_b128 %0, %1, off"
                 :: "v"(laddr), "v"(gaddr) : "memory");
}
__device__ __forceinline__ void wait_asynccnt0() {
    asm volatile("s_wait_asynccnt 0x0" ::: "memory");
}

// A fragment (16x32 f16, ISA layout): lane l<16 -> row l, K {0..7,16..23};
// lane l+16 -> row l, K {8..15,24..31}.  base points at (row0, k0), ld in halves.
__device__ __forceinline__ v16h load_afrag(const _Float16* base, int ld, int lane) {
    const int row  = lane & 15;
    const int koff = (lane < 16) ? 0 : 8;
    const _Float16* p = base + (size_t)row * ld + koff;
    v8h lo = *(const v8h*)(p);
    v8h hi = *(const v8h*)(p + 16);
    return cat8(lo, hi);
}

// B fragment (32x16 f16, ISA layout): lane n<16 -> col n, K {0..7,16..23};
// lane n+16 -> col n, K {8..15,24..31}.  base points at (col0 row of W[N,K], k0).
__device__ __forceinline__ v16h load_bfrag(const _Float16* base, int ld, int lane) {
    const int col  = lane & 15;
    const int koff = (lane < 16) ? 0 : 8;
    const _Float16* p = base + (size_t)col * ld + koff;
    v8h lo = *(const v8h*)(p);
    v8h hi = *(const v8h*)(p + 16);
    return cat8(lo, hi);
}

// ---------------------------------------------------------------------------
// attn_bias[b,l,m] = gamma * sum_c raw_bias[b,l,m,c] * w[c]   (67 MB read)
// ---------------------------------------------------------------------------
__global__ __launch_bounds__(256) void k_bias(const float* __restrict__ raw,
                                              const float* __restrict__ w,
                                              const float* __restrict__ gamma,
                                              float* __restrict__ out, long n) {
    long i = (long)blockIdx.x * blockDim.x + threadIdx.x;
    if (i >= n) return;
    float4 r = ((const float4*)raw)[i];
    out[i] = gamma[0] * (r.x * w[0] + r.y * w[1] + r.z * w[2] + r.w * w[3]);
}

// f32 -> f16 (n multiple of 4)
__global__ __launch_bounds__(256) void k_cvt(const float* __restrict__ in,
                                             _Float16* __restrict__ out, long n4) {
    long i = (long)blockIdx.x * blockDim.x + threadIdx.x;
    if (i >= n4) return;
    float4 v = ((const float4*)in)[i];
    v4h h; h.x = (_Float16)v.x; h.y = (_Float16)v.y; h.z = (_Float16)v.z; h.w = (_Float16)v.w;
    ((v4h*)out)[i] = h;
}

// vT[b,h,d,m] = qkv[b,m, 2D + h*96 + d]   (B-frag friendly layout for P@V)
__global__ __launch_bounds__(256) void k_vt(const _Float16* __restrict__ qkv,
                                            _Float16* __restrict__ vT) {
    long i = (long)blockIdx.x * blockDim.x + threadIdx.x;  // B*H*96*L
    if (i >= (long)BB * NH * HD * LSEQ) return;
    int m = (int)(i & (LSEQ - 1));
    long r = i >> 10;
    int d = (int)(r % HD); r /= HD;
    int h = (int)(r % NH);
    int b = (int)(r / NH);
    vT[i] = qkv[((size_t)(b * LSEQ + m)) * TD + 2 * DM + h * HD + d];
}

// ---------------------------------------------------------------------------
// GEMM: out[M,N] = A[M,K](f16) @ W[N,K]^T(f16) + bias[N]; optional exact GELU.
// Block: 4 waves (128 thr), wave tile 32x64, block tile 64x128, BK=32.
// A/B tiles staged in LDS via GLOBAL_LOAD_ASYNC_TO_LDS_B128, double buffered:
//   issue(next) -> compute(cur from LDS) -> s_wait_asynccnt 0 -> barrier.
// ---------------------------------------------------------------------------
__global__ __launch_bounds__(128) void k_gemm(const _Float16* __restrict__ A,
                                              const _Float16* __restrict__ W,
                                              const float* __restrict__ bias,
                                              int M, int N, int K,
                                              float* __restrict__ outF,
                                              _Float16* __restrict__ outH, int gelu) {
    __shared__ __align__(16) _Float16 As[2][64][APAD];    // 10 KB
    __shared__ __align__(16) _Float16 Bs[2][128][APAD];   // 20 KB

    const int t      = threadIdx.x;
    const int lane   = t & 31;
    const int wave   = t >> 5;
    const int tileM0 = blockIdx.y * 64;
    const int tileN0 = blockIdx.x * 128;
    const int wM     = (wave >> 1) * 32;
    const int wN     = (wave & 1) * 64;

    // issue one k-step's DMA: A 64x32 halves (2 chunks/thread), B 128x32 (4/thread)
    auto issue = [&](int s, int kk) {
        for (int i = 0; i < 2; ++i) {
            int q = t + i * 128;                 // 256 chunks
            int r = q >> 2, c = (q & 3) * 8;
            async_ldscopy_b128(A + (size_t)(tileM0 + r) * K + kk + c, &As[s][r][c]);
        }
        for (int i = 0; i < 4; ++i) {
            int q = t + i * 128;                 // 512 chunks
            int r = q >> 2, c = (q & 3) * 8;
            async_ldscopy_b128(W + (size_t)(tileN0 + r) * K + kk + c, &Bs[s][r][c]);
        }
    };

    v8f zero = {};
    v8f acc[2][4];
    for (int mh = 0; mh < 2; ++mh)
        for (int nh = 0; nh < 4; ++nh) acc[mh][nh] = zero;

    issue(0, 0);
    wait_asynccnt0();
    __syncthreads();

    const int row   = lane & 15;
    const int koffA = (lane < 16) ? 0 : 8;
    const int koffB = (lane < 16) ? 0 : 8;
    const int nsteps = K >> 5;

    for (int step = 0; step < nsteps; ++step) {
        const int s = step & 1;
        if (step + 1 < nsteps) issue(s ^ 1, (step + 1) << 5);

        v16h a0, a1, bf[4];
        {
            const _Float16* p0 = &As[s][wM + row][koffA];
            a0 = cat8(*(const v8h*)p0, *(const v8h*)(p0 + 16));
            const _Float16* p1 = &As[s][wM + 16 + row][koffA];
            a1 = cat8(*(const v8h*)p1, *(const v8h*)(p1 + 16));
            for (int nh = 0; nh < 4; ++nh) {
                const _Float16* pb = &Bs[s][wN + nh * 16 + row][koffB];
                bf[nh] = cat8(*(const v8h*)pb, *(const v8h*)(pb + 16));
            }
        }
        for (int nh = 0; nh < 4; ++nh) {
            acc[0][nh] = wmma_f16(a0, bf[nh], acc[0][nh]);
            acc[1][nh] = wmma_f16(a1, bf[nh], acc[1][nh]);
        }

        wait_asynccnt0();      // next stage's DMA landed in LDS
        __syncthreads();       // all waves done reading stage s
    }

    const int col0   = lane & 15;
    const int rowoff = (lane < 16) ? 0 : 8;
    for (int mh = 0; mh < 2; ++mh) {
        for (int nh = 0; nh < 4; ++nh) {
            const int col = tileN0 + wN + nh * 16 + col0;
            const float bv = bias ? bias[col] : 0.f;
            for (int r = 0; r < 8; ++r) {
                float v = acc[mh][nh][r] + bv;
                if (gelu) v = 0.5f * v * (1.f + erff(v * 0.70710678118654752f));
                size_t idx = (size_t)(tileM0 + wM + mh * 16 + rowoff + r) * N + col;
                if (outF) outF[idx] = v;
                if (outH) outH[idx] = (_Float16)v;
            }
        }
    }
}

// ---------------------------------------------------------------------------
// Flash attention: grid (L/64, H, B), 4 waves/block, 16 query rows per wave.
// scores = (Q K^T)/sqrt(96) + attn_bias; online softmax; O = P V.
// ---------------------------------------------------------------------------
__global__ __launch_bounds__(128) void k_attn(const _Float16* __restrict__ qkv,
                                              const _Float16* __restrict__ vT,
                                              const float* __restrict__ bias,
                                              _Float16* __restrict__ outH) {
    __shared__ __align__(16) _Float16 psh[4][16][APAD];   // per-wave P tile

    const int lane   = threadIdx.x & 31;
    const int wave   = threadIdx.x >> 5;
    const int b      = blockIdx.z;
    const int h      = blockIdx.y;
    const int qbase  = blockIdx.x * 64 + wave * 16;
    const int col0   = lane & 15;
    const int rowoff = (lane < 16) ? 0 : 8;
    const int koffB  = (lane < 16) ? 0 : 8;
    const float scale = 0.1020620726159658f;              // 1/sqrt(96)

    // Q fragments: A[16 x 96] -> 3 chunks of K=32
    v16h qa[3];
    const _Float16* qptr = qkv + ((size_t)(b * LSEQ + qbase)) * TD + h * HD;
    for (int c = 0; c < 3; ++c) qa[c] = load_afrag(qptr + c * 32, TD, lane);

    v8f zero = {};
    v8f o[6];
    for (int tt = 0; tt < 6; ++tt) o[tt] = zero;
    float mrow[8], lrow[8];
    for (int r = 0; r < 8; ++r) { mrow[r] = -1e30f; lrow[r] = 0.f; }

    const float* bptr = bias + (size_t)b * LSEQ * LSEQ;

    for (int j = 0; j < LSEQ; j += 32) {
        float S[2][8];
        for (int tt = 0; tt < 2; ++tt) {
            v8f s = zero;
            const _Float16* kptr =
                qkv + ((size_t)(b * LSEQ + j + tt * 16)) * TD + DM + h * HD;
            for (int c = 0; c < 3; ++c)
                s = wmma_f16(qa[c], load_bfrag(kptr + c * 32, TD, lane), s);
            const int kcol = j + tt * 16 + col0;
            for (int r = 0; r < 8; ++r)
                S[tt][r] = s[r] * scale +
                           bptr[(size_t)(qbase + rowoff + r) * LSEQ + kcol];
        }

        // online softmax statistics (rows live across 16-lane halves)
        float pA[8], pB[8], scl[8];
        for (int r = 0; r < 8; ++r) {
            float mx = fmaxf(S[0][r], S[1][r]);
            for (int m = 1; m < 16; m <<= 1) mx = fmaxf(mx, __shfl_xor(mx, m, 32));
            float nm = fmaxf(mrow[r], mx);
            scl[r]  = __expf(mrow[r] - nm);
            mrow[r] = nm;
            pA[r] = __expf(S[0][r] - nm);
            pB[r] = __expf(S[1][r] - nm);
            float sm = pA[r] + pB[r];
            for (int m = 1; m < 16; m <<= 1) sm += __shfl_xor(sm, m, 32);
            lrow[r] = lrow[r] * scl[r] + sm;
        }
        for (int tt = 0; tt < 6; ++tt)
            for (int r = 0; r < 8; ++r) o[tt][r] *= scl[r];

        // transpose P (C-layout) -> A-fragment layout via LDS
        for (int r = 0; r < 8; ++r) {
            psh[wave][rowoff + r][col0]      = (_Float16)pA[r];
            psh[wave][rowoff + r][16 + col0] = (_Float16)pB[r];
        }
        __syncthreads();
        v16h pa;
        {
            const int koff = (lane < 16) ? 0 : 8;
            const _Float16* pp = &psh[wave][col0][koff];
            pa = cat8(*(const v8h*)pp, *(const v8h*)(pp + 16));
        }
        __syncthreads();

        // O += P(16x32) @ V(32x96) using transposed-V layout
        for (int tt = 0; tt < 6; ++tt) {
            const _Float16* vp =
                vT + (((size_t)(b * NH + h) * HD) + tt * 16 + col0) * LSEQ + j + koffB;
            v16h vf = cat8(*(const v8h*)vp, *(const v8h*)(vp + 16));
            o[tt] = wmma_f16(pa, vf, o[tt]);
        }
    }

    for (int tt = 0; tt < 6; ++tt)
        for (int r = 0; r < 8; ++r) {
            float v = o[tt][r] / lrow[r];
            outH[((size_t)(b * LSEQ + qbase + rowoff + r)) * DM +
                 h * HD + tt * 16 + col0] = (_Float16)v;
        }
}

// ---------------------------------------------------------------------------
// Residual add + LayerNorm over D=768; emits f32 (residual chain) + f16 (GEMM A)
// ---------------------------------------------------------------------------
__global__ __launch_bounds__(256) void k_addln(const float* __restrict__ xin,
                                               const float* __restrict__ delta,
                                               const float* __restrict__ g,
                                               const float* __restrict__ bta,
                                               float* __restrict__ outF,
                                               _Float16* __restrict__ outH) {
    const int row  = blockIdx.x;
    const int t    = threadIdx.x;
    const int lane = t & 31;
    const int wave = t >> 5;
    __shared__ float wsum[8], wsum2[8];
    __shared__ float s_mu, s_rstd;

    const float* xr = xin   + (size_t)row * DM;
    const float* dr = delta + (size_t)row * DM;
    float v[3], s = 0.f, s2 = 0.f;
    for (int i = 0; i < 3; ++i) {
        int c = t + i * 256;
        v[i] = xr[c] + dr[c];
        s += v[i]; s2 += v[i] * v[i];
    }
    for (int m = 1; m < 32; m <<= 1) { s += __shfl_xor(s, m, 32); s2 += __shfl_xor(s2, m, 32); }
    if (lane == 0) { wsum[wave] = s; wsum2[wave] = s2; }
    __syncthreads();
    if (t == 0) {
        float ts = 0.f, ts2 = 0.f;
        for (int w = 0; w < 8; ++w) { ts += wsum[w]; ts2 += wsum2[w]; }
        float mu = ts * (1.f / DM);
        float var = ts2 * (1.f / DM) - mu * mu;
        s_mu = mu; s_rstd = rsqrtf(var + 1e-5f);
    }
    __syncthreads();
    const float mu = s_mu, rstd = s_rstd;
    for (int i = 0; i < 3; ++i) {
        int c = t + i * 256;
        float xv = (v[i] - mu) * rstd * g[c] + bta[c];
        outF[(size_t)row * DM + c] = xv;
        outH[(size_t)row * DM + c] = (_Float16)xv;
    }
}

// ---------------------------------------------------------------------------
extern "C" void kernel_launch(void* const* d_in, const int* in_sizes, int n_in,
                              void* d_out, int out_size, void* d_ws, size_t ws_size,
                              hipStream_t stream) {
    (void)in_sizes; (void)n_in; (void)out_size; (void)ws_size;

    const float* x_in   = (const float*)d_in[0];
    const float* rawb   = (const float*)d_in[1];
    const float* wmix   = (const float*)d_in[2];
    const float* gamma  = (const float*)d_in[3];
    const float* Wqkv   = (const float*)d_in[4];
    const float* bqkv   = (const float*)d_in[5];
    const float* Wo     = (const float*)d_in[6];
    const float* bo     = (const float*)d_in[7];
    const float* W1     = (const float*)d_in[8];
    const float* b1     = (const float*)d_in[9];
    const float* W2     = (const float*)d_in[10];
    const float* b2     = (const float*)d_in[11];
    const float* ln1g   = (const float*)d_in[12];
    const float* ln1b   = (const float*)d_in[13];
    const float* ln2g   = (const float*)d_in[14];
    const float* ln2b   = (const float*)d_in[15];
    float* out          = (float*)d_out;

    // ---- workspace carve (all sizes 256B-aligned by construction) ----
    char* p = (char*)d_ws;
    float*     biasF  = (float*)p;      p += (size_t)BB * LSEQ * LSEQ * 4;   // 16 MB
    _Float16*  xh     = (_Float16*)p;   p += (size_t)MROWS * DM * 2;         //  6 MB
    float*     xA     = (float*)p;      p += (size_t)MROWS * DM * 4;         // 12 MB
    float*     xB     = (float*)p;      p += (size_t)MROWS * DM * 4;         // 12 MB
    _Float16*  qkvH   = (_Float16*)p;   p += (size_t)MROWS * TD * 2;         // 18 MB
    _Float16*  vTH    = (_Float16*)p;   p += (size_t)BB * NH * HD * LSEQ * 2;//  6 MB
    _Float16*  attnH  = (_Float16*)p;   p += (size_t)MROWS * DM * 2;         //  6 MB
    float*     delta  = (float*)p;      p += (size_t)MROWS * DM * 4;         // 12 MB
    _Float16*  hH     = (_Float16*)p;   p += (size_t)MROWS * FF * 2;         // 16 MB
    _Float16*  WqkvH  = (_Float16*)p;   p += (size_t)2 * TD * DM * 2;
    _Float16*  WoH    = (_Float16*)p;   p += (size_t)2 * DM * DM * 2;
    _Float16*  W1H    = (_Float16*)p;   p += (size_t)2 * FF * DM * 2;
    _Float16*  W2H    = (_Float16*)p;   p += (size_t)2 * DM * FF * 2;

    const int TB = 256;
    auto g4 = [](long n4) { return (unsigned)((n4 + 255) / 256); };

    // weight + input conversions (weights stay L2-resident afterwards)
    k_cvt<<<g4((long)MROWS * DM / 4), TB, 0, stream>>>(x_in, xh, (long)MROWS * DM / 4);
    k_cvt<<<g4((long)2 * TD * DM / 4), TB, 0, stream>>>(Wqkv, WqkvH, (long)2 * TD * DM / 4);
    k_cvt<<<g4((long)2 * DM * DM / 4), TB, 0, stream>>>(Wo, WoH, (long)2 * DM * DM / 4);
    k_cvt<<<g4((long)2 * FF * DM / 4), TB, 0, stream>>>(W1, W1H, (long)2 * FF * DM / 4);
    k_cvt<<<g4((long)2 * DM * FF / 4), TB, 0, stream>>>(W2, W2H, (long)2 * DM * FF / 4);

    // structural bias (the single HBM-bound pass: 67 MB)
    k_bias<<<g4((long)BB * LSEQ * LSEQ), TB, 0, stream>>>(
        rawb, wmix, gamma, biasF, (long)BB * LSEQ * LSEQ);

    const float* xprev = x_in;
    for (int i = 0; i < 2; ++i) {
        // QKV projection -> f16 [B,L,3D]
        k_gemm<<<dim3(TD / 128, MROWS / 64), 128, 0, stream>>>(
            xh, WqkvH + (size_t)i * TD * DM, bqkv + (size_t)i * TD,
            MROWS, TD, DM, nullptr, qkvH, 0);
        // transpose V for P@V fragments
        k_vt<<<g4((long)BB * NH * HD * LSEQ), TB, 0, stream>>>(qkvH, vTH);
        // flash attention -> f16 [B,L,D]
        k_attn<<<dim3(LSEQ / 64, NH, BB), 128, 0, stream>>>(qkvH, vTH, biasF, attnH);
        // output projection -> f32 delta
        k_gemm<<<dim3(DM / 128, MROWS / 64), 128, 0, stream>>>(
            attnH, WoH + (size_t)i * DM * DM, bo + (size_t)i * DM,
            MROWS, DM, DM, delta, nullptr, 0);
        // x = LN(x + attn_out)
        k_addln<<<MROWS, TB, 0, stream>>>(xprev, delta, ln1g + (size_t)i * DM,
                                          ln1b + (size_t)i * DM, xA, xh);
        // FFN1 + exact GELU -> f16 hidden
        k_gemm<<<dim3(FF / 128, MROWS / 64), 128, 0, stream>>>(
            xh, W1H + (size_t)i * FF * DM, b1 + (size_t)i * FF,
            MROWS, FF, DM, nullptr, hH, 1);
        // FFN2 -> f32 delta
        k_gemm<<<dim3(DM / 128, MROWS / 64), 128, 0, stream>>>(
            hH, W2H + (size_t)i * DM * FF, b2 + (size_t)i * DM,
            MROWS, DM, FF, delta, nullptr, 0);
        // x = LN(x + ff); final layer writes straight to d_out
        float* lnout = (i == 1) ? out : xB;
        k_addln<<<MROWS, TB, 0, stream>>>(xA, delta, ln2g + (size_t)i * DM,
                                          ln2b + (size_t)i * DM, lnout, xh);
        xprev = lnout;
    }
}
